// RNNLayer_11965778887313
// MI455X (gfx1250) — matrix operand order repaired and checked by reading, not compiled, not run on hardware
//
#include <hip/hip_runtime.h>

typedef __attribute__((ext_vector_type(16))) __bf16 v16bf;
typedef __attribute__((ext_vector_type(8)))  float  v8f;

#define Bb   512
#define Nn   512
#define FINq 2
#define Ff   128
#define G4   512
#define MT   16     // batch tile per workgroup (one WMMA M tile)
#define LSTR 136    // LDS row stride in bf16 elems; 272B = multiple of 16B

union AOp { v16bf v; uint4 q[2]; };

// ---- activations: prefer gfx1250 hardware v_tanh_f32 -------------------
#if __has_builtin(__builtin_amdgcn_tanhf)
__device__ __forceinline__ float fast_tanh(float x) {
    return __builtin_amdgcn_tanhf(x);
}
__device__ __forceinline__ float fast_sigmoid(float x) {
    // sigmoid(x) = 0.5*tanh(0.5x) + 0.5  (1 TRANS + mul + fma)
    return __builtin_fmaf(0.5f, __builtin_amdgcn_tanhf(0.5f * x), 0.5f);
}
#elif __has_builtin(__builtin_amdgcn_rcpf)
__device__ __forceinline__ float fast_sigmoid(float x) {
    return __builtin_amdgcn_rcpf(1.0f + __expf(-x));   // v_rcp_f32, no div expansion
}
__device__ __forceinline__ float fast_tanh(float x) {
    return 2.0f * fast_sigmoid(2.0f * x) - 1.0f;
}
#else
__device__ __forceinline__ float fast_sigmoid(float x) {
    return 1.0f / (1.0f + __expf(-x));
}
__device__ __forceinline__ float fast_tanh(float x) {
    return 2.0f * fast_sigmoid(2.0f * x) - 1.0f;
}
#endif

#if __has_builtin(__builtin_amdgcn_s_barrier_signal) && \
    __has_builtin(__builtin_amdgcn_s_barrier_wait)
#define SPLIT_BARRIER 1
#else
#define SPLIT_BARRIER 0
#endif

__global__ void __launch_bounds__(256)
lstm_scan_kernel(const float* __restrict__ xin,   // (B, N, FIN) f32
                 const float* __restrict__ Wk,    // (FIN+F, 4F) f32
                 const float* __restrict__ bias,  // (4F)       f32
                 float* __restrict__ out)         // (B, N, F)  f32
{
    __shared__ __align__(16) __bf16 hbuf[2][MT][LSTR];

    const int tid  = threadIdx.x;
    const int wave = tid >> 5;          // 0..7  -> hidden chunk
    const int lane = tid & 31;
    const int b0   = blockIdx.x * MT;   // batch tile origin
    const int j0   = wave << 4;         // hidden units j0..j0+15
    const int n    = lane & 15;         // tile column / A row (m)
    const int lh   = lane >> 4;         // lane half
    const int offA = lh ? 16 : 0;       // A-layout K offset for this half

    // zero h double-buffer (h0 == 0)
    for (int i = tid; i < 2 * MT * LSTR; i += 256)
        ((__bf16*)hbuf)[i] = (__bf16)0.0f;

    // ---- preload weight B-operands: 4 gates x 5 K-slices, resident in VGPRs
    // A-K space: [h(0..127) | x0(128) x1(129) | one(130) | zero-pad(..159)]
    // B k order used here (contiguous): col = lane%16; lanes 0-15 hold K 0..15, lanes 16-31 K 16..31.
    v16bf Bop[4][5];
    #pragma unroll
    for (int g = 0; g < 4; ++g) {
        const int col = g * Ff + j0 + n;
        #pragma unroll
        for (int ks = 0; ks < 5; ++ks) {
            const int kbase = ks * 32 + lh * 16;
            v16bf bb;
            #pragma unroll
            for (int e = 0; e < 16; ++e) {
                const int k = kbase + e;
                float w = 0.0f;
                if (k < Ff)               w = Wk[(size_t)(FINq + k) * G4 + col]; // Wh
                else if (k - Ff < FINq)   w = Wk[(size_t)(k - Ff) * G4 + col];   // Wx
                else if (k == Ff + FINq)  w = bias[col];                          // bias row
                bb[e] = (__bf16)w;
            }
            Bop[g][ks] = bb;
        }
    }

    float cst[8];                        // cell state, C/D layout
    #pragma unroll
    for (int i = 0; i < 8; ++i) cst[i] = 0.0f;

    // persistent x/bias A-slice (slice 4): only elems 0,1 change per step
    v16bf a4;
    #pragma unroll
    for (int e = 0; e < 16; ++e) a4[e] = (__bf16)0.0f;
    if (lh == 0) a4[2] = (__bf16)1.0f;   // K=130: multiplies the bias row

    float cx0 = 0.0f, cx1 = 0.0f;        // x for current step (exclusive: 0 at step 0)

    __syncthreads();

    #pragma unroll 1
    for (int step = 0; step < Nn; ++step) {
        const int buf = step & 1;

        // ---- prefetch x for NEXT step (inputs[:, step]) a full iteration early
        float nx0 = 0.0f, nx1 = 0.0f;
        if (lh == 0 && step + 1 < Nn) {
            const float* xp = xin + ((size_t)(b0 + n) * Nn + step) * FINq;
            nx0 = xp[0];
            nx1 = xp[1];
        }

        // ---- A operands: h from LDS (slices 0..3)
        // A k order used here (contiguous, same as B): m = lane%16; lanes 0-15 K 0..15, lanes 16-31 K 16..31
        v16bf A[4];
        #pragma unroll
        for (int ks = 0; ks < 4; ++ks) {
            const __bf16* p = &hbuf[buf][n][ks * 32 + offA];
            AOp a;
            a.q[0] = *(const uint4*)p;          // K offA .. offA+7
            a.q[1] = *(const uint4*)(p + 8);    // K offA+8 .. offA+15
            A[ks] = a.v;
        }
        a4[0] = (__bf16)cx0;   // K=128
        a4[1] = (__bf16)cx1;   // K=129

        // ---- gates: acc[g] = [x|1] @ [Wx;bias] (C = inline 0) then += h @ Wh
        v8f acc[4];
        #pragma unroll
        for (int g = 0; g < 4; ++g) {
            v8f z;
            #pragma unroll
            for (int i = 0; i < 8; ++i) z[i] = 0.0f;
            v8f a = __builtin_amdgcn_wmma_f32_16x16x32_bf16(
                        false, a4, false, Bop[g][4], (short)0, z, false, false);
            #pragma unroll
            for (int ks = 0; ks < 4; ++ks)
                a = __builtin_amdgcn_wmma_f32_16x16x32_bf16(
                        false, A[ks], false, Bop[g][ks], (short)0, a, false, false);
            acc[g] = a;
        }

        // ---- LSTM cell update; stage h to LDS for next step
        const int nbuf = buf ^ 1;
        float hv[8];
        #pragma unroll
        for (int i = 0; i < 8; ++i) {
            const float ig = fast_sigmoid(acc[0][i]);
            const float fg = fast_sigmoid(acc[1][i]);
            const float gg = fast_tanh(acc[2][i]);
            const float og = fast_sigmoid(acc[3][i]);
            cst[i] = fg * cst[i] + ig * gg;
            hv[i] = og * fast_tanh(cst[i]);
            const int mm = i + (lh << 3);   // D layout row
            hbuf[nbuf][mm][j0 + n] = (__bf16)hv[i];
        }

#if SPLIT_BARRIER
        // make LDS h visible, arrive at barrier, then overlap the global
        // stores (fire-and-forget) with the other waves' arrival.
        asm volatile("s_wait_dscnt 0x0" ::: "memory");
        __builtin_amdgcn_s_barrier_signal(-1);
        #pragma unroll
        for (int i = 0; i < 8; ++i) {
            const int mm = i + (lh << 3);
            __builtin_nontemporal_store(
                hv[i], &out[((size_t)(b0 + mm) * Nn + step) * Ff + j0 + n]);
        }
        cx0 = nx0; cx1 = nx1;
        __builtin_amdgcn_s_barrier_wait((short)-1);
        asm volatile("" ::: "memory");
#else
        #pragma unroll
        for (int i = 0; i < 8; ++i) {
            const int mm = i + (lh << 3);
            __builtin_nontemporal_store(
                hv[i], &out[((size_t)(b0 + mm) * Nn + step) * Ff + j0 + n]);
        }
        cx0 = nx0; cx1 = nx1;
        __syncthreads();
#endif
    }
}

extern "C" void kernel_launch(void* const* d_in, const int* in_sizes, int n_in,
                              void* d_out, int out_size, void* d_ws, size_t ws_size,
                              hipStream_t stream) {
    const float* xin  = (const float*)d_in[0];   // inputs (B,N,FIN)
    const float* Wk   = (const float*)d_in[1];   // (FIN+F, 4F)
    const float* bias = (const float*)d_in[2];   // (4F)
    float* out = (float*)d_out;                  // (B,N,F)
    dim3 grid(Bb / MT), block(256);
    hipLaunchKernelGGL(lstm_scan_kernel, grid, block, 0, stream, xin, Wk, bias, out);
}
